// Qwen3MLP_20383914787225
// MI455X (gfx1250) — matrix operand order repaired and checked by reading, not compiled, not run on hardware
//
#include <hip/hip_runtime.h>
#include <hip/hip_bf16.h>

// ---------------------------------------------------------------------------
// SwiGLU MLP for MI455X (gfx1250, wave32, WMMA + async-to-LDS).
//   g = x @ gate_w^T ; u = x @ up_w^T ; h = silu(g)*u ; out = h @ down_w^T
// Compute-bound ~30x (2.47 TFLOP vs ~36us HBM at 23.3 TB/s) => bf16 WMMA
// (16x16x32, f32 accum; 8x the K-depth of f32 WMMA). Operands pre-converted
// to bf16 once; bf16 gate+up weights (192 MB) fit the 192 MB L2 in GEMM1.
// Staging uses GLOBAL_LOAD_ASYNC_TO_LDS_B128 (ASYNCcnt) with double-buffered
// LDS: async loads of slab s+1 overlap WMMA of slab s; one s_wait_asynccnt
// + one barrier per slab.
// ---------------------------------------------------------------------------

#define T_DIM 8192
#define H_DIM 4096
#define I_DIM 12288

typedef __attribute__((ext_vector_type(16))) __bf16 v16bf;
typedef __attribute__((ext_vector_type(8)))  __bf16 v8bf;
typedef __attribute__((ext_vector_type(4)))  __bf16 v4bf;
typedef __attribute__((ext_vector_type(8)))  float  v8f;

// LDS tile: 128 rows x 64 K, padded row stride (72 elem = 144 B, 16B aligned)
#define LDS_STRIDE 72
#define TILE_ELEMS (128 * LDS_STRIDE)

// ---------------------------------------------------------------------------
// CDNA5 async copy: global -> LDS, 16B per lane, tracked by ASYNCcnt.
//   vdst = per-lane LDS byte address, vaddr = per-lane 32-bit byte offset,
//   saddr = uniform 64-bit base (GVS mode).
// ---------------------------------------------------------------------------
__device__ __forceinline__ void async_b128(unsigned lds_byte,
                                           const void* base,
                                           unsigned byte_off) {
  asm volatile("global_load_async_to_lds_b128 %0, %1, %2"
               :
               : "v"(lds_byte), "v"(byte_off), "s"(base)
               : "memory");
}

__device__ __forceinline__ void wait_async0() {
  asm volatile("s_wait_asynccnt 0" ::: "memory");
}

// Low 32 bits of a flat pointer into the shared aperture == LDS byte offset.
__device__ __forceinline__ unsigned lds_off(const void* p) {
  return (unsigned)(unsigned long long)p;
}

// ---------------------------------------------------------------------------
// fp32 -> bf16 conversion, 4 elements/thread/iter (float4 in, 8B out)
// ---------------------------------------------------------------------------
__global__ __launch_bounds__(256) void cvt_f32_to_bf16(
    const float* __restrict__ in, __bf16* __restrict__ out, long n) {
  long i = ((long)blockIdx.x * 256 + threadIdx.x) * 4;
  const long stride = (long)gridDim.x * 256 * 4;
  for (; i < n; i += stride) {
    float4 v = *(const float4*)(in + i);
    v4bf o;
    o[0] = (__bf16)v.x; o[1] = (__bf16)v.y;
    o[2] = (__bf16)v.z; o[3] = (__bf16)v.w;
    *(v4bf*)(out + i) = o;
  }
}

// ---------------------------------------------------------------------------
// Fragment loaders (CDNA5 WMMA register layouts, ISA 7.12.2).
// A (16x32 bf16): lanes 0-15 hold M=lane, K in {0..7, 16..23};
//                 lanes 16-31 hold M=lane-16, K in {8..15, 24..31}.
// B (32x16 bf16): lanes 0-15 hold N=lane, K in {0..7, 16..23};
//                 lanes 16-31 hold N=lane-16, K in {8..15, 24..31}.
// ---------------------------------------------------------------------------
__device__ __forceinline__ v16bf load_a_frag(const __bf16* s, int rowBase,
                                             int kk, int lane) {
  const int m  = rowBase + (lane & 15);
  const int kb = (lane >> 4) * 8;
  const __bf16* p = s + m * LDS_STRIDE + kk + kb;
  v8bf lo = *(const v8bf*)p;          // K = kb .. kb+7
  v8bf hi = *(const v8bf*)(p + 16);   // K = 16+kb .. 16+kb+7
  v16bf r;
#pragma unroll
  for (int j = 0; j < 8; ++j) { r[j] = lo[j]; r[j + 8] = hi[j]; }
  return r;
}

__device__ __forceinline__ v16bf load_b_frag(const __bf16* s, int colBase,
                                             int kk, int lane) {
  const int n  = colBase + (lane & 15);
  const int kb = (lane >> 4) * 8;
  const __bf16* p = s + n * LDS_STRIDE + kk + kb;
  v8bf lo = *(const v8bf*)p;          // K = kb .. kb+7
  v8bf hi = *(const v8bf*)(p + 16);   // K = 16+kb .. 16+kb+7
  v16bf r;
#pragma unroll
  for (int j = 0; j < 8; ++j) { r[j] = lo[j]; r[j + 8] = hi[j]; }
  return r;
}

#define WMMA_BF16(A, B, C) \
  __builtin_amdgcn_wmma_f32_16x16x32_bf16(false, (A), false, (B), (short)0, \
                                          (C), false, false)

// ---------------------------------------------------------------------------
// GEMM1 (fused gate+up+silu):  h[T,I] = silu(x @ Wg^T) * (x @ Wu^T)
// Block: 256 thr (8 waves), tile 128(M) x 128(N), BK=64.
// Waves: 4(M-groups of 32) x 2(N-groups of 64); each wave: 2x4 acc tiles per
// matmul (16 v8f accs total). Double-buffered LDS, async global->LDS staging.
// ---------------------------------------------------------------------------
__global__ __launch_bounds__(256) void swiglu_gemm1(
    const __bf16* __restrict__ X, const __bf16* __restrict__ Wg,
    const __bf16* __restrict__ Wu, __bf16* __restrict__ Hout) {
  __shared__ __bf16 sA[2][TILE_ELEMS];
  __shared__ __bf16 sG[2][TILE_ELEMS];
  __shared__ __bf16 sU[2][TILE_ELEMS];

  const int tid  = threadIdx.x;
  const int lane = tid & 31;
  const int wave = tid >> 5;
  const int wm   = wave & 3;   // M group: 32*wm
  const int wn   = wave >> 2;  // N group: 64*wn
  const long m0  = (long)blockIdx.y * 128;
  const long n0  = (long)blockIdx.x * 128;

  v8f accG[2][4];
  v8f accU[2][4];
#pragma unroll
  for (int mt = 0; mt < 2; ++mt)
#pragma unroll
    for (int nt = 0; nt < 4; ++nt) {
      accG[mt][nt] = (v8f)(0.0f);
      accU[mt][nt] = (v8f)(0.0f);
    }

  // Tile = 128 rows x 64 cols bf16 = 1024 x 16B chunks, 4 chunks/thread/op.
  unsigned ldsChunk[4];   // LDS byte offset of chunk within a tile buffer
  unsigned offA[4], offW[4];  // global byte offsets at k=0 (A rows / W rows)
#pragma unroll
  for (int i = 0; i < 4; ++i) {
    const int c   = tid + i * 256;
    const int row = c >> 3;
    const int col = (c & 7) * 8;
    ldsChunk[i] = (unsigned)((row * LDS_STRIDE + col) * 2);
    offA[i] = (unsigned)(((m0 + row) * H_DIM + col) * 2);
    offW[i] = (unsigned)(((n0 + row) * H_DIM + col) * 2);
  }
  const unsigned ldsA[2] = {lds_off(&sA[0][0]), lds_off(&sA[1][0])};
  const unsigned ldsG[2] = {lds_off(&sG[0][0]), lds_off(&sG[1][0])};
  const unsigned ldsU[2] = {lds_off(&sU[0][0]), lds_off(&sU[1][0])};

  auto issue = [&](int stage, int buf) {
    const unsigned kb = (unsigned)(stage * 64 * 2);
#pragma unroll
    for (int i = 0; i < 4; ++i) {
      async_b128(ldsA[buf] + ldsChunk[i], X,  offA[i] + kb);
      async_b128(ldsG[buf] + ldsChunk[i], Wg, offW[i] + kb);
      async_b128(ldsU[buf] + ldsChunk[i], Wu, offW[i] + kb);
    }
  };

  issue(0, 0);
  const int S = H_DIM / 64;
#pragma unroll 1
  for (int s = 0; s < S; ++s) {
    wait_async0();      // my async writes to buf (s&1) have landed
    __syncthreads();    // => everyone's have; and buf ((s+1)&1) is free
    if (s + 1 < S) issue(s + 1, (s + 1) & 1);
    const int buf = s & 1;
    const __bf16* pA = &sA[buf][0];
    const __bf16* pG = &sG[buf][0];
    const __bf16* pU = &sU[buf][0];

#pragma unroll
    for (int kk = 0; kk < 64; kk += 32) {
      v16bf a[2];
#pragma unroll
      for (int mt = 0; mt < 2; ++mt)
        a[mt] = load_a_frag(pA, wm * 32 + mt * 16, kk, lane);
#pragma unroll
      for (int nt = 0; nt < 4; ++nt) {
        const v16bf bg = load_b_frag(pG, wn * 64 + nt * 16, kk, lane);
        const v16bf bu = load_b_frag(pU, wn * 64 + nt * 16, kk, lane);
#pragma unroll
        for (int mt = 0; mt < 2; ++mt) {
          accG[mt][nt] = WMMA_BF16(a[mt], bg, accG[mt][nt]);
          accU[mt][nt] = WMMA_BF16(a[mt], bu, accU[mt][nt]);
        }
      }
    }
  }

  // Epilogue: h = g*sigmoid(g)*u, stored bf16.
  // C/D layout: VGPR r -> M = r + (lane>=16 ? 8 : 0), N = lane&15.
  const int hi8   = (lane >> 4) * 8;
  const int nlane = lane & 15;
#pragma unroll
  for (int mt = 0; mt < 2; ++mt)
#pragma unroll
    for (int nt = 0; nt < 4; ++nt) {
      const long col = n0 + wn * 64 + nt * 16 + nlane;
#pragma unroll
      for (int r = 0; r < 8; ++r) {
        const float g = accG[mt][nt][r];
        const float u = accU[mt][nt][r];
        const float h = g * u * __builtin_amdgcn_rcpf(1.0f + __expf(-g));
        const long row = m0 + wm * 32 + mt * 16 + r + hi8;
        Hout[row * (long)I_DIM + col] = (__bf16)h;
      }
    }
}

// ---------------------------------------------------------------------------
// GEMM2: out[T,H] = h @ Wd^T   (f32 output)
// ---------------------------------------------------------------------------
__global__ __launch_bounds__(256) void swiglu_gemm2(
    const __bf16* __restrict__ Hin, const __bf16* __restrict__ Wd,
    float* __restrict__ Out) {
  __shared__ __bf16 sA[2][TILE_ELEMS];
  __shared__ __bf16 sB[2][TILE_ELEMS];

  const int tid  = threadIdx.x;
  const int lane = tid & 31;
  const int wave = tid >> 5;
  const int wm   = wave & 3;
  const int wn   = wave >> 2;
  const long m0  = (long)blockIdx.y * 128;
  const long n0  = (long)blockIdx.x * 128;

  v8f acc[2][4];
#pragma unroll
  for (int mt = 0; mt < 2; ++mt)
#pragma unroll
    for (int nt = 0; nt < 4; ++nt) acc[mt][nt] = (v8f)(0.0f);

  unsigned ldsChunk[4];
  unsigned offA[4], offW[4];
#pragma unroll
  for (int i = 0; i < 4; ++i) {
    const int c   = tid + i * 256;
    const int row = c >> 3;
    const int col = (c & 7) * 8;
    ldsChunk[i] = (unsigned)((row * LDS_STRIDE + col) * 2);
    offA[i] = (unsigned)(((m0 + row) * I_DIM + col) * 2);
    offW[i] = (unsigned)(((n0 + row) * I_DIM + col) * 2);
  }
  const unsigned ldsA[2] = {lds_off(&sA[0][0]), lds_off(&sA[1][0])};
  const unsigned ldsB[2] = {lds_off(&sB[0][0]), lds_off(&sB[1][0])};

  auto issue = [&](int stage, int buf) {
    const unsigned kb = (unsigned)(stage * 64 * 2);
#pragma unroll
    for (int i = 0; i < 4; ++i) {
      async_b128(ldsA[buf] + ldsChunk[i], Hin, offA[i] + kb);
      async_b128(ldsB[buf] + ldsChunk[i], Wd,  offW[i] + kb);
    }
  };

  issue(0, 0);
  const int S = I_DIM / 64;
#pragma unroll 1
  for (int s = 0; s < S; ++s) {
    wait_async0();
    __syncthreads();
    if (s + 1 < S) issue(s + 1, (s + 1) & 1);
    const int buf = s & 1;
    const __bf16* pA = &sA[buf][0];
    const __bf16* pB = &sB[buf][0];

#pragma unroll
    for (int kk = 0; kk < 64; kk += 32) {
      v16bf a[2];
#pragma unroll
      for (int mt = 0; mt < 2; ++mt)
        a[mt] = load_a_frag(pA, wm * 32 + mt * 16, kk, lane);
#pragma unroll
      for (int nt = 0; nt < 4; ++nt) {
        const v16bf b = load_b_frag(pB, wn * 64 + nt * 16, kk, lane);
#pragma unroll
        for (int mt = 0; mt < 2; ++mt)
          acc[mt][nt] = WMMA_BF16(a[mt], b, acc[mt][nt]);
      }
    }
  }

  const int hi8   = (lane >> 4) * 8;
  const int nlane = lane & 15;
#pragma unroll
  for (int mt = 0; mt < 2; ++mt)
#pragma unroll
    for (int nt = 0; nt < 4; ++nt) {
      const long col = n0 + wn * 64 + nt * 16 + nlane;
#pragma unroll
      for (int r = 0; r < 8; ++r) {
        const long row = m0 + wm * 32 + mt * 16 + r + hi8;
        Out[row * (long)H_DIM + col] = acc[mt][nt][r];
      }
    }
}

// ---------------------------------------------------------------------------
// Launch. Workspace layout (bf16):
//   xb [T*H] | gb [I*H] | ub [I*H] | db [H*I] | hb [T*I]   (~544 MB total)
// ---------------------------------------------------------------------------
extern "C" void kernel_launch(void* const* d_in, const int* in_sizes, int n_in,
                              void* d_out, int out_size, void* d_ws,
                              size_t ws_size, hipStream_t stream) {
  const float* x  = (const float*)d_in[0];
  const float* gw = (const float*)d_in[1];
  const float* uw = (const float*)d_in[2];
  const float* dw = (const float*)d_in[3];
  float* out = (float*)d_out;

  __bf16* xb = (__bf16*)d_ws;
  __bf16* gb = xb + (size_t)T_DIM * H_DIM;
  __bf16* ub = gb + (size_t)I_DIM * H_DIM;
  __bf16* db = ub + (size_t)I_DIM * H_DIM;
  __bf16* hb = db + (size_t)H_DIM * I_DIM;

  cvt_f32_to_bf16<<<2048, 256, 0, stream>>>(x,  xb, (long)T_DIM * H_DIM);
  cvt_f32_to_bf16<<<2048, 256, 0, stream>>>(gw, gb, (long)I_DIM * H_DIM);
  cvt_f32_to_bf16<<<2048, 256, 0, stream>>>(uw, ub, (long)I_DIM * H_DIM);
  cvt_f32_to_bf16<<<2048, 256, 0, stream>>>(dw, db, (long)H_DIM * I_DIM);

  dim3 g1(I_DIM / 128, T_DIM / 128);  // 96 x 64
  swiglu_gemm1<<<g1, 256, 0, stream>>>(xb, gb, ub, hb);

  dim3 g2(H_DIM / 128, T_DIM / 128);  // 32 x 64
  swiglu_gemm2<<<g2, 256, 0, stream>>>(hb, db, out);
}
